// GATLayer_42339787604598
// MI455X (gfx1250) — matrix operand order repaired and checked by reading, not compiled, not run on hardware
//
#include <hip/hip_runtime.h>
#include <hip/hip_bf16.h>

// ---------------- problem constants ----------------
constexpr int Bn = 4, Nn = 2048, Cc = 3, Hh = 4, Dd = 32;
constexpr int IN_DIM = 128, OUT_DIM = 128;
constexpr float NEG_INF = -1000000000.0f;

typedef __attribute__((ext_vector_type(16))) _Float16 v16h;
typedef __attribute__((ext_vector_type(8)))  _Float16 v8h;
typedef __attribute__((ext_vector_type(8)))  float    v8f;
typedef __attribute__((ext_vector_type(4)))  float    v4f;

union U16H { v16h v; v8h h2[2]; };
union PK2  { _Float16 h[2]; unsigned u; };

// spread 8 bits so bit m lands at position 4m
__device__ __forceinline__ unsigned spread4(unsigned x) {
    x = (x | (x << 12)) & 0x000F000Fu;
    x = (x | (x << 6))  & 0x03030303u;
    x = (x | (x << 3))  & 0x11111111u;
    return x;
}

// ---------------- kernel 0: W / Wo -> transposed f16 ----------------
__global__ void k_prep(const float* __restrict__ W, const float* __restrict__ Wo,
                       _Float16* __restrict__ WT, _Float16* __restrict__ WoT) {
    int t = blockIdx.x * blockDim.x + threadIdx.x;   // 16384
    int i = t >> 7, o = t & 127;
    WT [o * 128 + i] = (_Float16)W [i * 128 + o];
    WoT[o * 128 + i] = (_Float16)Wo[i * 128 + o];
}

// ---------------- kernel 1: adjacency -> packed mask bits ----------------
// One wave covers 128 consecutive j for a fixed (b,i). Per lane: one float4
// per channel, nontemporal (adj is read exactly once; keep it out of L2's
// hot set). Ballot per sub-element, then lanes 0..3 re-pack into bit words.
// maskbits[b][i][j/32] bit j%32 = (sum over c of adj > 0)  <=> mean > 0
__global__ void k_mask(const float* __restrict__ adj, unsigned* __restrict__ mb) {
    int wid  = blockIdx.x * (blockDim.x >> 5) + (threadIdx.x >> 5); // 131072 waves
    int lane = threadIdx.x & 31;
    int j0 = (wid & 15) * 128;          // 16 j-chunks per row
    int i  = (wid >> 4) & (Nn - 1);
    int b  = wid >> 15;
    const long long nn = (long long)Nn * Nn;
    const float* p = adj + (long long)b * Cc * nn + (long long)i * Nn + j0 + lane * 4;
    v4f s0 = __builtin_nontemporal_load((const v4f*)p);
    v4f s1 = __builtin_nontemporal_load((const v4f*)(p + nn));
    v4f s2 = __builtin_nontemporal_load((const v4f*)(p + 2 * nn));
    v4f s;
    for (int q = 0; q < 4; ++q) s[q] = s0[q] + s1[q] + s2[q];
    unsigned bal0 = (unsigned)__ballot(s[0] > 0.0f);
    unsigned bal1 = (unsigned)__ballot(s[1] > 0.0f);
    unsigned bal2 = (unsigned)__ballot(s[2] > 0.0f);
    unsigned bal3 = (unsigned)__ballot(s[3] > 0.0f);
    if (lane < 4) {
        int sh = lane * 8;
        unsigned word = spread4((bal0 >> sh) & 0xFFu)
                      | (spread4((bal1 >> sh) & 0xFFu) << 1)
                      | (spread4((bal2 >> sh) & 0xFFu) << 2)
                      | (spread4((bal3 >> sh) & 0xFFu) << 3);
        mb[((long long)(b * Nn + i)) * 64 + (j0 >> 5) + lane] = word;
    }
}

// ---------------- kernel 2: Wh = h @ W (WMMA f16->f32) ----------------
// writes Wh f32 [b*N][128] and WhT f16 [b][h][d][n]
__global__ void k_wh(const float* __restrict__ h, const _Float16* __restrict__ WT,
                     float* __restrict__ Wh, _Float16* __restrict__ WhT) {
    int wave = threadIdx.x >> 5, lane = threadIdx.x & 31;
    int tile = blockIdx.x * (blockDim.x >> 5) + wave;   // 512 tiles
    int row0 = tile * 16;
    int r = lane & 15, hi = lane >> 4;

    v8f acc[8] = {};
    for (int k = 0; k < IN_DIM; k += 32) {
        // A fragment: h rows row0..row0+15, K slice in contiguous order k = 16*hi + s
        const float* ap = h + (long long)(row0 + r) * IN_DIM + k + hi * 16;
        v4f a0 = *(const v4f*)(ap);
        v4f a1 = *(const v4f*)(ap + 4);
        v4f a2 = *(const v4f*)(ap + 8);
        v4f a3 = *(const v4f*)(ap + 12);
        v16h A;
        for (int q = 0; q < 4; ++q) {
            A[q]      = (_Float16)a0[q];
            A[4 + q]  = (_Float16)a1[q];
            A[8 + q]  = (_Float16)a2[q];
            A[12 + q] = (_Float16)a3[q];
        }
        for (int nt = 0; nt < 8; ++nt) {
            v16h Bf = *(const v16h*)(WT + (nt * 16 + r) * 128 + k + hi * 16);
            acc[nt] = __builtin_amdgcn_wmma_f32_16x16x32_f16(
                false, A, false, Bf, (short)0, acc[nt], false, false);
        }
    }
    int b = row0 >> 11;               // row0 / N
    int nbase = row0 & (Nn - 1);
    for (int v = 0; v < 8; ++v) {
        int rr = v + hi * 8;
        long long grow = row0 + rr;
        for (int nt = 0; nt < 8; ++nt) {
            int col = nt * 16 + r;
            float val = acc[nt][v];
            Wh[grow * 128 + col] = val;
            int hd = col >> 5, d = col & 31;
            WhT[(((long long)b * Hh + hd) * Dd + d) * Nn + (nbase + rr)] = (_Float16)val;
        }
    }
}

// ---------------- kernel 3: src/dst = Wh . a_src/a_dst ----------------
__global__ void k_srcdst(const float* __restrict__ Wh, const float* __restrict__ a_src,
                         const float* __restrict__ a_dst, float* __restrict__ srcv,
                         float* __restrict__ dstv) {
    int t = blockIdx.x * blockDim.x + threadIdx.x;   // B*N*H = 32768
    int hd = t & 3, n = (t >> 2) & (Nn - 1), b = t >> 13;
    const float* wp = Wh + ((long long)(b * Nn + n)) * 128 + hd * Dd;
    const float* as = a_src + hd * Dd;
    const float* ad = a_dst + hd * Dd;
    float s = 0.f, d = 0.f;
    for (int q = 0; q < Dd; ++q) { float w = wp[q]; s += w * as[q]; d += w * ad[q]; }
    int idx = (b * Hh + hd) * Nn + n;
    srcv[idx] = s; dstv[idx] = d;
}

// ---------------- kernel 4: per-row softmax stats (max, 1/sum) ----------------
__global__ void k_rowstats(const float* __restrict__ srcv, const float* __restrict__ dstv,
                           const unsigned* __restrict__ mb, float* __restrict__ rowm,
                           float* __restrict__ rowis) {
    __shared__ float dsh[Nn];
    int bh = blockIdx.x >> 8;                 // (b*H+h), 256 blocks each
    int i0 = (blockIdx.x & 255) * 8;
    int b = bh >> 2;
    for (int q = threadIdx.x; q < Nn; q += 256) dsh[q] = dstv[bh * Nn + q];
    __syncthreads();
    int wave = threadIdx.x >> 5, lane = threadIdx.x & 31;
    int i = i0 + wave;
    float sv = srcv[bh * Nn + i];
    const unsigned* mrow = mb + ((long long)(b * Nn + i)) * 64;
    float mx = NEG_INF;
    for (int w = 0; w < 64; ++w) {
        unsigned word = mrow[w];
        float e = sv + dsh[w * 32 + lane];
        e = (e > 0.f) ? e : 0.2f * e;
        e = ((word >> lane) & 1u) ? e : NEG_INF;
        mx = fmaxf(mx, e);
    }
    for (int off = 16; off; off >>= 1) mx = fmaxf(mx, __shfl_xor(mx, off, 32));
    float sum = 0.f;
    for (int w = 0; w < 64; ++w) {
        unsigned word = mrow[w];
        float e = sv + dsh[w * 32 + lane];
        e = (e > 0.f) ? e : 0.2f * e;
        e = ((word >> lane) & 1u) ? e : NEG_INF;
        sum += __expf(e - mx);
    }
    for (int off = 16; off; off >>= 1) sum += __shfl_xor(sum, off, 32);
    if (lane == 0) { rowm[bh * Nn + i] = mx; rowis[bh * Nn + i] = 1.0f / sum; }
}

// ---------------- kernel 5: fused softmax @ Wh (flash-style, WMMA) ----------------
// block = 128 threads (4 waves = 4 heads), one 16-row i-tile per block
__global__ void k_attn(const float* __restrict__ srcv, const float* __restrict__ dstv,
                       const float* __restrict__ rowm, const float* __restrict__ rowis,
                       const unsigned* __restrict__ mb, const _Float16* __restrict__ WhT,
                       _Float16* __restrict__ attn) {
    __shared__ unsigned msh[16][64];                      // 4 KB mask bits for 16 rows
    __shared__ __align__(32) _Float16 Psh[4][16 * 32];    // 4 KB P tiles (per wave)
    __shared__ float dsh[4][Nn];                          // 32 KB dst rows (per head)

    int b  = blockIdx.x >> 7;          // 128 i-tiles per batch
    int i0 = (blockIdx.x & 127) * 16;
    int wave = threadIdx.x >> 5;       // head
    int lane = threadIdx.x & 31;
    int bh = b * Hh + wave;
    int r = lane & 15, hi = lane >> 4;

    for (int q = threadIdx.x; q < 16 * 64; q += 128) {
        int rr = q >> 6, ww = q & 63;
        msh[rr][ww] = mb[((long long)(b * Nn + i0 + rr)) * 64 + ww];
    }
    for (int q = lane; q < Nn; q += 32) dsh[wave][q] = dstv[bh * Nn + q];
    __syncthreads();

    float sv = srcv [bh * Nn + i0 + r];
    float mv = rowm [bh * Nn + i0 + r];
    float is = rowis[bh * Nn + i0 + r];
    _Float16* Pw = &Psh[wave][0];
    const float* dl = &dsh[wave][0];

    v8f acc0 = {}, acc1 = {};
    for (int j0 = 0; j0 < Nn; j0 += 32) {
        unsigned word = msh[r][j0 >> 5];
        int cbase = hi * 16;
        for (int cc = 0; cc < 16; cc += 2) {
            float e0 = sv + dl[j0 + cbase + cc];
            float e1 = sv + dl[j0 + cbase + cc + 1];
            e0 = (e0 > 0.f) ? e0 : 0.2f * e0;
            e1 = (e1 > 0.f) ? e1 : 0.2f * e1;
            e0 = ((word >> (cbase + cc))     & 1u) ? e0 : NEG_INF;
            e1 = ((word >> (cbase + cc + 1)) & 1u) ? e1 : NEG_INF;
            PK2 pk;
            pk.h[0] = (_Float16)(__expf(e0 - mv) * is);
            pk.h[1] = (_Float16)(__expf(e1 - mv) * is);
            *(unsigned*)(Pw + r * 32 + cbase + cc) = pk.u;
        }
        __syncthreads();   // uniform trip count across all 4 waves

        // A fragment from LDS: contiguous k = 16*hi + s
        U16H ua;
        ua.h2[0] = *(const v8h*)(Pw + r * 32 + hi * 16);
        ua.h2[1] = *(const v8h*)(Pw + r * 32 + hi * 16 + 8);
        // B fragments: contiguous 32B per lane from WhT[b][h][d][j]
        const _Float16* b0p = WhT + ((long long)(bh * Dd + r))      * Nn + j0 + hi * 16;
        const _Float16* b1p = WhT + ((long long)(bh * Dd + 16 + r)) * Nn + j0 + hi * 16;
        v16h B0 = *(const v16h*)b0p;
        v16h B1 = *(const v16h*)b1p;
        acc0 = __builtin_amdgcn_wmma_f32_16x16x32_f16(false, ua.v, false, B0, (short)0, acc0, false, false);
        acc1 = __builtin_amdgcn_wmma_f32_16x16x32_f16(false, ua.v, false, B1, (short)0, acc1, false, false);
    }
    for (int v = 0; v < 8; ++v) {
        int rr = v + hi * 8;
        long long row = (long long)b * Nn + i0 + rr;
        attn[row * 128 + wave * Dd + r]      = (_Float16)acc0[v];
        attn[row * 128 + wave * Dd + 16 + r] = (_Float16)acc1[v];
    }
}

// ---------------- kernel 6: out = elu(attn @ Wo + bo) ----------------
__global__ void k_out(const _Float16* __restrict__ attn, const _Float16* __restrict__ WoT,
                      const float* __restrict__ bo, float* __restrict__ out) {
    int wave = threadIdx.x >> 5, lane = threadIdx.x & 31;
    int tile = blockIdx.x * (blockDim.x >> 5) + wave;   // 512 tiles
    long long row0 = (long long)tile * 16;
    int r = lane & 15, hi = lane >> 4;

    v8f acc[8] = {};
    for (int k = 0; k < OUT_DIM; k += 32) {
        const _Float16* ap = attn + (row0 + r) * 128 + k + hi * 16;
        U16H ua;
        ua.h2[0] = *(const v8h*)(ap);
        ua.h2[1] = *(const v8h*)(ap + 8);
        for (int nt = 0; nt < 8; ++nt) {
            v16h Bf = *(const v16h*)(WoT + (nt * 16 + r) * 128 + k + hi * 16);
            acc[nt] = __builtin_amdgcn_wmma_f32_16x16x32_f16(
                false, ua.v, false, Bf, (short)0, acc[nt], false, false);
        }
    }
    for (int nt = 0; nt < 8; ++nt) {
        int col = nt * 16 + r;
        float bias = bo[col];
        for (int v = 0; v < 8; ++v) {
            float x = acc[nt][v] + bias;
            x = (x > 0.f) ? x : (__expf(x) - 1.0f);
            out[(row0 + v + hi * 8) * 128 + col] = x;
        }
    }
}

// ---------------- launch ----------------
extern "C" void kernel_launch(void* const* d_in, const int* in_sizes, int n_in,
                              void* d_out, int out_size, void* d_ws, size_t ws_size,
                              hipStream_t stream) {
    (void)in_sizes; (void)n_in; (void)out_size; (void)ws_size;
    const float* h     = (const float*)d_in[0];
    const float* adj   = (const float*)d_in[1];
    const float* W     = (const float*)d_in[2];
    const float* a_src = (const float*)d_in[3];
    const float* a_dst = (const float*)d_in[4];
    const float* Wo    = (const float*)d_in[5];
    const float* bo    = (const float*)d_in[6];
    float* out = (float*)d_out;

    char* ws = (char*)d_ws;
    _Float16* WT    = (_Float16*)(ws + 0);          //  32 KB
    _Float16* WoT   = (_Float16*)(ws + 32768);      //  32 KB
    float*    Wh    = (float*)   (ws + 65536);      //   4 MB
    _Float16* WhT   = (_Float16*)(ws + 4259840);    //   2 MB
    float*    srcv  = (float*)   (ws + 6356992);    // 128 KB
    float*    dstv  = (float*)   (ws + 6488064);    // 128 KB
    unsigned* mbits = (unsigned*)(ws + 6619136);    //   2 MB
    float*    rowm  = (float*)   (ws + 8716288);    // 128 KB
    float*    rowis = (float*)   (ws + 8847360);    // 128 KB
    _Float16* attn  = (_Float16*)(ws + 8978432);    //   2 MB

    k_prep    <<<64, 256, 0, stream>>>(W, Wo, WT, WoT);
    k_mask    <<<(Bn * Nn * (Nn / 128)) / 8, 256, 0, stream>>>(adj, mbits);
    k_wh      <<<128, 128, 0, stream>>>(h, WT, Wh, WhT);
    k_srcdst  <<<128, 256, 0, stream>>>(Wh, a_src, a_dst, srcv, dstv);
    k_rowstats<<<(Bn * Hh * Nn) / 8, 256, 0, stream>>>(srcv, dstv, mbits, rowm, rowis);
    k_attn    <<<Bn * (Nn / 16), 128, 0, stream>>>(srcv, dstv, rowm, rowis, mbits, WhT, attn);
    k_out     <<<128, 128, 0, stream>>>(attn, WoT, bo, out);
}
